// Model_82291573392145
// MI455X (gfx1250) — matrix operand order repaired and checked by reading, not compiled, not run on hardware
//
#include <hip/hip_runtime.h>
#include <math.h>

// ---------------- types for WMMA (gfx1250, wave32) ----------------
typedef __attribute__((ext_vector_type(16))) __bf16 v16bf;
typedef __attribute__((ext_vector_type(8)))  __bf16 v8bf;
typedef __attribute__((ext_vector_type(8)))  float  v8f;

// Problem constants
#define BB   8
#define NN   500
#define NNP  512            // padded per-batch rows for adj slabs
#define QQ   25
#define ENC  512
#define DIN  300
#define DINP 320
#define NC_  70
#define NET  3
#define MTOT (BB*NN)        // 4000
#define MPAD 4032           // 63 * 64  (block M-tile multiple)

// ---------------------------------------------------------------
// Fragment loaders (no bounds checks: all A operands are zero-padded
// to a multiple of the 64-row block tile).
// A (16x32 bf16, row-major, ISA layout): lane = g*16+mi, element e maps to
// K = (e/8)*16 + g*8 + (e%8)  => two contiguous 16-byte chunks per lane.
// B (32x16 bf16) from n-major (transposed) Bt[n*ldb + k]:
// lane mi selects column n, element e maps to K = (e/8)*16 + g*8 + (e%8) (same as A).
// ---------------------------------------------------------------
__device__ __forceinline__ v16bf load_afrag(const __bf16* __restrict__ A, int lda,
                                            int row, int k0, int g) {
  union U { v16bf v; v8bf h[2]; } u;
  const __bf16* p = A + (size_t)row * lda + k0 + g * 8;
  u.h[0] = *(const v8bf*)(p);
  u.h[1] = *(const v8bf*)(p + 16);
  return u.v;
}

__device__ __forceinline__ v16bf load_bfrag(const __bf16* __restrict__ Bt, int ldb,
                                            int col, int k0, int g) {
  union U { v16bf v; v8bf h[2]; } u;
  const __bf16* p = Bt + (size_t)col * ldb + k0 + g * 8;
  u.h[0] = *(const v8bf*)(p);
  u.h[1] = *(const v8bf*)(p + 16);
  return u.v;
}

// One wave computes a 32x32 tile: acc[2][2] of 16x16 fragments.
__device__ __forceinline__ void gemm_tile(const __bf16* __restrict__ A, int lda,
                                          const __bf16* __restrict__ Bt, int ldb,
                                          int K, int wm, int wn, v8f (&acc)[2][2]) {
  const int lane = threadIdx.x & 31;
  const int g = lane >> 4, mi = lane & 15;
  for (int k0 = 0; k0 < K; k0 += 32) {
    v16bf a0 = load_afrag(A, lda, wm + mi,      k0, g);
    v16bf a1 = load_afrag(A, lda, wm + 16 + mi, k0, g);
    v16bf b0 = load_bfrag(Bt, ldb, wn + mi,      k0, g);
    v16bf b1 = load_bfrag(Bt, ldb, wn + 16 + mi, k0, g);
    acc[0][0] = __builtin_amdgcn_wmma_f32_16x16x32_bf16(false, a0, false, b0, (short)0, acc[0][0], false, false);
    acc[0][1] = __builtin_amdgcn_wmma_f32_16x16x32_bf16(false, a0, false, b1, (short)0, acc[0][1], false, false);
    acc[1][0] = __builtin_amdgcn_wmma_f32_16x16x32_bf16(false, a1, false, b0, (short)0, acc[1][0], false, false);
    acc[1][1] = __builtin_amdgcn_wmma_f32_16x16x32_bf16(false, a1, false, b1, (short)0, acc[1][1], false, false);
  }
}

#define GEMM_PROLOG()                                             \
  const int lane = threadIdx.x & 31, wave = threadIdx.x >> 5;     \
  const int wm = blockIdx.y * 64 + (wave >> 2) * 32;              \
  const int wn = blockIdx.x * 128 + (wave & 3) * 32;              \
  const int g = lane >> 4, mi = lane & 15;                        \
  v8f acc[2][2];                                                  \
  _Pragma("unroll") for (int i = 0; i < 2; ++i)                   \
  _Pragma("unroll") for (int j = 0; j < 2; ++j)                   \
  _Pragma("unroll") for (int e = 0; e < 8; ++e) acc[i][j][e] = 0.0f;

// -------------------- prep kernels --------------------
__global__ void k_transpose_pad_bf16(const float* __restrict__ W, __bf16* __restrict__ Wt,
                                     int K, int N, int Kpad) {
  int idx = blockIdx.x * 256 + threadIdx.x;
  if (idx >= N * Kpad) return;
  int n = idx / Kpad, k = idx % Kpad;
  Wt[idx] = (k < K) ? (__bf16)W[(size_t)k * N + n] : (__bf16)0.0f;
}

// row/col padded convert: Y is rows x Kpad, X is srcRows x K
__global__ void k_convert_pad(const float* __restrict__ X, __bf16* __restrict__ Y,
                              int rows, int srcRows, int K, int Kpad) {
  int idx = blockIdx.x * 256 + threadIdx.x;
  if (idx >= rows * Kpad) return;
  int r = idx / Kpad, k = idx % Kpad;
  Y[idx] = (r < srcRows && k < K) ? (__bf16)X[(size_t)r * K + k] : (__bf16)0.0f;
}

// adj: (B*NET) slabs of 500x500 -> slabs of 512x512, zero padded
__global__ void k_convert_adj(const float* __restrict__ X, __bf16* __restrict__ Y) {
  int idx = blockIdx.x * 256 + threadIdx.x;
  if (idx >= BB * NET * NNP * NNP) return;
  int j = idx & (NNP - 1);
  int row = idx >> 9;          // slab*512 + i
  int slab = row >> 9, i = row & (NNP - 1);
  float v = 0.0f;
  if (i < NN && j < NN) v = X[((size_t)slab * NN + i) * NN + j];
  Y[idx] = (__bf16)v;
}

__global__ void k_zero_bf16(__bf16* __restrict__ p, int n) {
  int idx = blockIdx.x * 256 + threadIdx.x;
  if (idx < n) p[idx] = (__bf16)0.0f;
}

// -------------------- GEMM kernels --------------------
// nodes_compress = tanh(nodes@Wn + bn); last_hop = nc * nmask (f32 + bf16)
__global__ __launch_bounds__(256) void k_gemm_nc(const __bf16* __restrict__ A, const __bf16* __restrict__ Bt,
                                                 const float* __restrict__ bias, const int* __restrict__ nlen,
                                                 float* __restrict__ nc, float* __restrict__ lh,
                                                 __bf16* __restrict__ lhb) {
  GEMM_PROLOG();
  gemm_tile(A, DINP, Bt, DINP, DINP, wm, wn, acc);
#pragma unroll
  for (int i = 0; i < 2; ++i)
#pragma unroll
    for (int j = 0; j < 2; ++j)
#pragma unroll
      for (int e = 0; e < 8; ++e) {
        int row = wm + i * 16 + g * 8 + e;
        if (row >= MTOT) continue;
        int col = wn + j * 16 + mi;
        float v = tanhf(acc[i][j][e] + bias[col]);
        size_t idx = (size_t)row * ENC + col;
        nc[idx] = v;
        int bb = row / NN, ii = row - bb * NN;
        float l = (ii < nlen[bb]) ? v : 0.0f;
        lh[idx] = l;
        lhb[idx] = (__bf16)l;
      }
}

// hm = (last_hop@Wh + bh) * nmask ; also write transposed bf16 copy per batch
__global__ __launch_bounds__(256) void k_gemm_hm(const __bf16* __restrict__ A, const __bf16* __restrict__ Bt,
                                                 const float* __restrict__ bias, const int* __restrict__ nlen,
                                                 float* __restrict__ hm, __bf16* __restrict__ hmT) {
  GEMM_PROLOG();
  gemm_tile(A, ENC, Bt, ENC, ENC, wm, wn, acc);
#pragma unroll
  for (int i = 0; i < 2; ++i)
#pragma unroll
    for (int j = 0; j < 2; ++j)
#pragma unroll
      for (int e = 0; e < 8; ++e) {
        int row = wm + i * 16 + g * 8 + e;
        if (row >= MTOT) continue;
        int col = wn + j * 16 + mi;
        int bb = row / NN, ii = row - bb * NN;
        float h = (ii < nlen[bb]) ? (acc[i][j][e] + bias[col]) : 0.0f;
        hm[(size_t)row * ENC + col] = h;
        hmT[(size_t)bb * ENC * ENC + (size_t)col * ENC + ii] = (__bf16)h;
      }
}

// update = sum_e adj[b,e]@hm[b] + hm ; pack catA = [update | last_hop] in bf16
__global__ __launch_bounds__(256) void k_gemm_upd(const __bf16* __restrict__ adjb, const __bf16* __restrict__ hmT,
                                                  const float* __restrict__ hm, const float* __restrict__ lh,
                                                  float* __restrict__ upd, __bf16* __restrict__ catA) {
  const int b = blockIdx.z;
  GEMM_PROLOG();
  const __bf16* BT = hmT + (size_t)b * ENC * ENC;
  for (int e = 0; e < NET; ++e) {
    const __bf16* A = adjb + ((size_t)(b * NET + e) * NNP) * NNP;
    gemm_tile(A, NNP, BT, ENC, NNP, wm, wn, acc);
  }
#pragma unroll
  for (int i = 0; i < 2; ++i)
#pragma unroll
    for (int j = 0; j < 2; ++j)
#pragma unroll
      for (int e = 0; e < 8; ++e) {
        int r = wm + i * 16 + g * 8 + e;
        if (r >= NN) continue;
        int col = wn + j * 16 + mi;
        int grow = b * NN + r;
        size_t idx = (size_t)grow * ENC + col;
        float u = acc[i][j][e] + hm[idx];
        upd[idx] = u;
        catA[(size_t)grow * (2 * ENC) + col] = (__bf16)u;
        catA[(size_t)grow * (2 * ENC) + ENC + col] = (__bf16)lh[idx];
      }
}

// att = sigmoid(cat@Wc + bc)*nmask ; last_hop = att*tanh(update)+(1-att)*last_hop
__global__ __launch_bounds__(256) void k_gemm_att(const __bf16* __restrict__ A, const __bf16* __restrict__ Bt,
                                                  const float* __restrict__ bias, const int* __restrict__ nlen,
                                                  const float* __restrict__ upd, float* __restrict__ lh,
                                                  __bf16* __restrict__ lhb) {
  GEMM_PROLOG();
  gemm_tile(A, 2 * ENC, Bt, 2 * ENC, 2 * ENC, wm, wn, acc);
#pragma unroll
  for (int i = 0; i < 2; ++i)
#pragma unroll
    for (int j = 0; j < 2; ++j)
#pragma unroll
      for (int e = 0; e < 8; ++e) {
        int row = wm + i * 16 + g * 8 + e;
        if (row >= MTOT) continue;
        int col = wn + j * 16 + mi;
        size_t idx = (size_t)row * ENC + col;
        int bb = row / NN, ii = row - bb * NN;
        float s = acc[i][j][e] + bias[col];
        float att = (ii < nlen[bb]) ? (1.0f / (1.0f + expf(-s))) : 0.0f;
        float u = upd[idx], l = lh[idx];
        float nl = att * tanhf(u) + (1.0f - att) * l;
        lh[idx] = nl;
        lhb[idx] = (__bf16)nl;
      }
}

// hidden = tanh(g@W1 + b1), N=128
__global__ __launch_bounds__(256) void k_gemm_hidden(const __bf16* __restrict__ A, const __bf16* __restrict__ Bt,
                                                     const float* __restrict__ bias, float* __restrict__ hidden) {
  GEMM_PROLOG();
  gemm_tile(A, 4 * ENC, Bt, 4 * ENC, 4 * ENC, wm, wn, acc);
#pragma unroll
  for (int i = 0; i < 2; ++i)
#pragma unroll
    for (int j = 0; j < 2; ++j)
#pragma unroll
      for (int e = 0; e < 8; ++e) {
        int row = wm + i * 16 + g * 8 + e;
        if (row >= MTOT) continue;
        int col = wn + j * 16 + mi;
        hidden[(size_t)row * 128 + col] = tanhf(acc[i][j][e] + bias[col]);
      }
}

// -------------------- small fp32 kernels --------------------
__global__ void k_query(const float* __restrict__ qg, const float* __restrict__ Wq,
                        const float* __restrict__ bq, float* __restrict__ qc) {
  int idx = blockIdx.x * 256 + threadIdx.x;
  if (idx >= BB * QQ * ENC) return;
  int d = idx & (ENC - 1);
  int r = idx >> 9;
  const float* x = qg + (size_t)r * DIN;
  float s = bq[d];
  for (int k = 0; k < DIN; ++k) s += x[k] * Wq[(size_t)k * ENC + d];
  qc[idx] = s;
}

__global__ void k_aq(const float* __restrict__ qc, const float* __restrict__ wa,
                     float* __restrict__ aq) {
  int i = blockIdx.x * 256 + threadIdx.x;
  if (i >= BB * QQ) return;
  float s = 0.0f;
  const float* q = qc + (size_t)i * ENC;
  for (int d = 0; d < ENC; ++d) s += q[d] * wa[ENC + d];
  aq[i] = s;
}

// per node row: sim row (Q=25), softmax over Q, nodes2query, rowmax
__global__ void k_sim(const float* __restrict__ lh, const float* __restrict__ qc,
                      const float* __restrict__ wa, const float* __restrict__ aq,
                      float* __restrict__ rowmax, float* __restrict__ n2q) {
  __shared__ float Lws[ENC];
  __shared__ float red[256];
  __shared__ float s25[32];
  __shared__ float p25[32];
  __shared__ float an_sh;
  int row = blockIdx.x;
  int b = row / NN;
  int tid = threadIdx.x;
  const float* L = lh + (size_t)row * ENC;
  float l0 = L[tid], l1 = L[tid + 256];
  Lws[tid] = l0 * wa[2 * ENC + tid];
  Lws[tid + 256] = l1 * wa[2 * ENC + tid + 256];
  red[tid] = l0 * wa[tid] + l1 * wa[tid + 256];
  __syncthreads();
  for (int s = 128; s > 0; s >>= 1) { if (tid < s) red[tid] += red[tid + s]; __syncthreads(); }
  if (tid == 0) an_sh = red[0];
  __syncthreads();
  int wv = tid >> 5, lane = tid & 31;
  const float* qcb = qc + (size_t)b * QQ * ENC;
  for (int q = wv; q < QQ; q += 8) {
    const float* qrow = qcb + (size_t)q * ENC;
    float part = 0.0f;
    for (int d = lane; d < ENC; d += 32) part += Lws[d] * qrow[d];
#pragma unroll
    for (int off = 16; off > 0; off >>= 1) part += __shfl_xor(part, off, 32);
    if (lane == 0) s25[q] = an_sh + aq[b * QQ + q] + part;
  }
  __syncthreads();
  if (tid == 0) {
    float m = s25[0];
    for (int q = 1; q < QQ; ++q) m = fmaxf(m, s25[q]);
    float sum = 0.0f;
    for (int q = 0; q < QQ; ++q) { float e = expf(s25[q] - m); p25[q] = e; sum += e; }
    float inv = 1.0f / sum;
    for (int q = 0; q < QQ; ++q) p25[q] *= inv;
    rowmax[row] = m;
  }
  __syncthreads();
  float a0 = 0.0f, a1 = 0.0f;
  for (int q = 0; q < QQ; ++q) {
    float p = p25[q];
    const float* qrow = qcb + (size_t)q * ENC;
    a0 += p * qrow[tid];
    a1 += p * qrow[tid + 256];
  }
  n2q[(size_t)row * ENC + tid] = a0;
  n2q[(size_t)row * ENC + tid + 256] = a1;
}

// per batch: bvec = softmax over N of rowmax ; q2n = bvec @ nodes_compress
__global__ void k_bvec(const float* __restrict__ rowmax, const float* __restrict__ nc,
                       float* __restrict__ q2n) {
  __shared__ float red[256];
  __shared__ float bv[NN];
  __shared__ float mx_sh, inv_sh;
  int b = blockIdx.x, tid = threadIdx.x;
  float lm = -3.4e38f;
  for (int n = tid; n < NN; n += 256) lm = fmaxf(lm, rowmax[b * NN + n]);
  red[tid] = lm; __syncthreads();
  for (int s = 128; s > 0; s >>= 1) { if (tid < s) red[tid] = fmaxf(red[tid], red[tid + s]); __syncthreads(); }
  if (tid == 0) mx_sh = red[0];
  __syncthreads();
  float ls = 0.0f;
  for (int n = tid; n < NN; n += 256) { float e = expf(rowmax[b * NN + n] - mx_sh); bv[n] = e; ls += e; }
  red[tid] = ls; __syncthreads();
  for (int s = 128; s > 0; s >>= 1) { if (tid < s) red[tid] += red[tid + s]; __syncthreads(); }
  if (tid == 0) inv_sh = 1.0f / red[0];
  __syncthreads();
  float a0 = 0.0f, a1 = 0.0f;
  for (int n = 0; n < NN; ++n) {
    float p = bv[n];
    const float* r = nc + ((size_t)(b * NN + n)) * ENC;
    a0 += p * r[tid];
    a1 += p * r[tid + 256];
  }
  q2n[b * ENC + tid] = a0 * inv_sh;
  q2n[b * ENC + tid + 256] = a1 * inv_sh;
}

// build g over MPAD rows (zero pad rows) in bf16
__global__ void k_build_g(const float* __restrict__ nc, const float* __restrict__ n2q,
                          const float* __restrict__ q2n, __bf16* __restrict__ gbuf) {
  int idx = blockIdx.x * 256 + threadIdx.x;
  if (idx >= MPAD * 4 * ENC) return;
  int row = idx >> 11, j = idx & (4 * ENC - 1);
  if (row >= MTOT) { gbuf[idx] = (__bf16)0.0f; return; }
  int d = j & (ENC - 1), sec = j >> 9;
  int b = row / NN;
  float ncv = nc[(size_t)row * ENC + d];
  float v;
  if (sec == 0) v = ncv;
  else if (sec == 1) v = n2q[(size_t)row * ENC + d];
  else if (sec == 2) v = ncv * n2q[(size_t)row * ENC + d];
  else v = ncv * q2n[b * ENC + d];
  gbuf[idx] = (__bf16)v;
}

__global__ void k_raw(const float* __restrict__ hidden, const float* __restrict__ W2,
                      const float* __restrict__ b2, float* __restrict__ raw) {
  __shared__ float red[128];
  int row = blockIdx.x, tid = threadIdx.x;
  red[tid] = hidden[(size_t)row * 128 + tid] * W2[tid];
  __syncthreads();
  for (int s = 64; s > 0; s >>= 1) { if (tid < s) red[tid] += red[tid + s]; __syncthreads(); }
  if (tid == 0) raw[row] = red[0] + b2[0];
}

__global__ void k_pred(const int* __restrict__ mask, const float* __restrict__ raw,
                       float* __restrict__ out) {
  __shared__ float red[256];
  int bc = blockIdx.x;
  int b = bc / NC_, c = bc - b * NC_;
  int tid = threadIdx.x;
  float lm = -3.4e38f;
  const int* mrow = mask + ((size_t)b * NC_ + c) * NN;
  const float* rrow = raw + b * NN;
  for (int n = tid; n < NN; n += 256) {
    float pv = (float)mrow[n] * rrow[n];
    if (pv == 0.0f) pv = -1000000.0f;
    lm = fmaxf(lm, pv);
  }
  red[tid] = lm; __syncthreads();
  for (int s = 128; s > 0; s >>= 1) { if (tid < s) red[tid] = fmaxf(red[tid], red[tid + s]); __syncthreads(); }
  if (tid == 0) out[bc] = red[0];
}

// -------------------- host launcher --------------------
extern "C" void kernel_launch(void* const* d_in, const int* in_sizes, int n_in,
                              void* d_out, int out_size, void* d_ws, size_t ws_size,
                              hipStream_t stream) {
  (void)in_sizes; (void)n_in; (void)out_size; (void)ws_size;
  const float* nodes_glove = (const float*)d_in[0];
  const float* query_glove = (const float*)d_in[1];
  const float* adj         = (const float*)d_in[2];
  const int*   nlen        = (const int*)  d_in[3];
  const int*   mask        = (const int*)  d_in[4];
  const float* Wn = (const float*)d_in[5];
  const float* bn = (const float*)d_in[6];
  const float* Wq = (const float*)d_in[7];
  const float* bq = (const float*)d_in[8];
  const float* Wh = (const float*)d_in[9];
  const float* bh = (const float*)d_in[10];
  const float* Wc = (const float*)d_in[11];
  const float* bc = (const float*)d_in[12];
  const float* wa = (const float*)d_in[13];
  const float* W1 = (const float*)d_in[14];
  const float* b1 = (const float*)d_in[15];
  const float* W2 = (const float*)d_in[16];
  const float* b2 = (const float*)d_in[17];
  float* out = (float*)d_out;

  char* base = (char*)d_ws;
  size_t off = 0;
  auto ALLOC = [&](size_t bytes) -> void* {
    void* p = base + off;
    off += (bytes + 511) & ~(size_t)511;
    return p;
  };

  float*  nc_f  = (float*) ALLOC((size_t)MTOT * ENC * 4);
  float*  lh_f  = (float*) ALLOC((size_t)MTOT * ENC * 4);
  float*  upd_f = (float*) ALLOC((size_t)MTOT * ENC * 4);
  float*  hm_f  = (float*) ALLOC((size_t)MTOT * ENC * 4);   // reused as n2q
  float*  qc_f  = (float*) ALLOC((size_t)BB * QQ * ENC * 4);
  __bf16* lhb   = (__bf16*)ALLOC((size_t)MPAD * ENC * 2);
  __bf16* hmT   = (__bf16*)ALLOC((size_t)BB * ENC * ENC * 2);
  // region R: catA (MPAD x 1024 bf16) + adjb (B*NET slabs of 512x512 bf16),
  // reused after the hops as g (MPAD x 2048 bf16)
  size_t catA_bytes = (size_t)MPAD * 2 * ENC * 2;
  size_t adjb_bytes = (size_t)BB * NET * NNP * NNP * 2;
  char*   R     = (char*)  ALLOC(catA_bytes + adjb_bytes);
  __bf16* catA  = (__bf16*)R;
  __bf16* adjb  = (__bf16*)(R + catA_bytes);
  __bf16* gb    = (__bf16*)R;
  __bf16* ngb   = (__bf16*)ALLOC((size_t)MPAD * DINP * 2);
  __bf16* WnT   = (__bf16*)ALLOC((size_t)ENC * DINP * 2);
  __bf16* WhT   = (__bf16*)ALLOC((size_t)ENC * ENC * 2);
  __bf16* WcT   = (__bf16*)ALLOC((size_t)ENC * 2 * ENC * 2);
  __bf16* W1T   = (__bf16*)ALLOC((size_t)128 * 4 * ENC * 2);
  float*  aqv   = (float*) ALLOC((size_t)BB * QQ * 4);
  float*  rowmx = (float*) ALLOC((size_t)MTOT * 4);
  float*  q2n   = (float*) ALLOC((size_t)BB * ENC * 4);
  float*  raw   = (float*) ALLOC((size_t)MTOT * 4);
  float*  hidden = upd_f;   // upd dead after hops
  float*  n2q    = hm_f;    // hm dead after hops

  dim3 blk(256);

  // weight prep (bf16, transposed n-major, K padded to mult of 32)
  k_transpose_pad_bf16<<<(ENC * DINP + 255) / 256, blk, 0, stream>>>(Wn, WnT, DIN, ENC, DINP);
  k_transpose_pad_bf16<<<(ENC * ENC + 255) / 256, blk, 0, stream>>>(Wh, WhT, ENC, ENC, ENC);
  k_transpose_pad_bf16<<<(ENC * 2 * ENC + 255) / 256, blk, 0, stream>>>(Wc, WcT, 2 * ENC, ENC, 2 * ENC);
  k_transpose_pad_bf16<<<(128 * 4 * ENC + 255) / 256, blk, 0, stream>>>(W1, W1T, 4 * ENC, 128, 4 * ENC);
  // activation conversions (row + K padded so GEMM loops are guard-free)
  k_convert_pad<<<(MPAD * DINP + 255) / 256, blk, 0, stream>>>(nodes_glove, ngb, MPAD, MTOT, DIN, DINP);
  k_convert_adj<<<(BB * NET * NNP * NNP + 255) / 256, blk, 0, stream>>>(adj, adjb);
  k_zero_bf16<<<(BB * ENC * ENC + 255) / 256, blk, 0, stream>>>(hmT, BB * ENC * ENC);
  // zero pad rows of lhb and catA (rows MTOT..MPAD-1), written once per launch
  k_zero_bf16<<<((MPAD - MTOT) * ENC + 255) / 256, blk, 0, stream>>>(lhb + (size_t)MTOT * ENC, (MPAD - MTOT) * ENC);
  k_zero_bf16<<<((MPAD - MTOT) * 2 * ENC + 255) / 256, blk, 0, stream>>>(catA + (size_t)MTOT * 2 * ENC, (MPAD - MTOT) * 2 * ENC);

  // encoder
  k_gemm_nc<<<dim3(ENC / 128, MPAD / 64), blk, 0, stream>>>(ngb, WnT, bn, nlen, nc_f, lh_f, lhb);
  k_query<<<(BB * QQ * ENC + 255) / 256, blk, 0, stream>>>(query_glove, Wq, bq, qc_f);

  // 3 hops
  for (int h = 0; h < 3; ++h) {
    k_gemm_hm<<<dim3(ENC / 128, MPAD / 64), blk, 0, stream>>>(lhb, WhT, bh, nlen, hm_f, hmT);
    k_gemm_upd<<<dim3(ENC / 128, NNP / 64, BB), blk, 0, stream>>>(adjb, hmT, hm_f, lh_f, upd_f, catA);
    k_gemm_att<<<dim3(ENC / 128, MPAD / 64), blk, 0, stream>>>(catA, WcT, bc, nlen, upd_f, lh_f, lhb);
  }

  // attention over queries
  k_aq<<<(BB * QQ + 255) / 256, blk, 0, stream>>>(qc_f, wa, aqv);
  k_sim<<<MTOT, blk, 0, stream>>>(lh_f, qc_f, wa, aqv, rowmx, n2q);
  k_bvec<<<BB, blk, 0, stream>>>(rowmx, nc_f, q2n);

  // scorer
  k_build_g<<<(MPAD * 4 * ENC + 255) / 256, blk, 0, stream>>>(nc_f, n2q, q2n, gb);
  k_gemm_hidden<<<dim3(1, MPAD / 64), blk, 0, stream>>>(gb, W1T, b1, hidden);
  k_raw<<<MTOT, dim3(128), 0, stream>>>(hidden, W2, b2, raw);
  k_pred<<<BB * NC_, blk, 0, stream>>>(mask, raw, out);
}
